// TransformerModel_13503377179143
// MI455X (gfx1250) — matrix operand order repaired and checked by reading, not compiled, not run on hardware
//
#include <hip/hip_runtime.h>
#include <math.h>

// ---------------------------------------------------------------------------
// Transformer encoder-decoder forward for MI455X (gfx1250), wave32 + WMMA.
// All GEMMs and attention matmuls use v_wmma_f32_16x16x32_bf16.
// GEMM A-tiles are staged with GLOBAL_LOAD_ASYNC_TO_LDS_B128 (ASYNCcnt path).
// ---------------------------------------------------------------------------

typedef __bf16 bf16_t;
typedef __attribute__((ext_vector_type(16))) __bf16 v16bf;
typedef __attribute__((ext_vector_type(8)))  float  v8f;

static constexpr int B_  = 8;
static constexpr int S_  = 1024;   // S == T
static constexpr int D_  = 512;
static constexpr int H_  = 8;
static constexpr int DK_ = 64;
static constexpr int FF_ = 2048;
static constexpr int L_  = 6;
static constexpr int M_  = B_ * S_;   // 8192 rows

union Frag {
    v16bf  v;
    uint4  u[2];
    bf16_t h[16];
};

static __device__ __forceinline__ bf16_t f2bf(float x) {
    unsigned u = __float_as_uint(x);
    u += 0x7fffu + ((u >> 16) & 1u);          // round-to-nearest-even
    unsigned short hs = (unsigned short)(u >> 16);
    bf16_t r;
    __builtin_memcpy(&r, &hs, 2);
    return r;
}

static __device__ __forceinline__ v8f wmma_bf16(v16bf a, v16bf b, v8f c) {
    return __builtin_amdgcn_wmma_f32_16x16x32_bf16(
        /*neg_a=*/false, a, /*neg_b=*/false, b,
        /*c_mod=*/(short)0, c, /*reuse_a=*/false, /*reuse_b=*/false);
}

// Async copy 32B global -> LDS as two 16B ops (VDST = LDS byte address, 64-bit VADDR, GV mode).
// IOFFSET applies to both LDS and global address (ISA 08_async_tensor §4.4).
static __device__ __forceinline__ void async_b128(unsigned lds_addr, const void* gaddr) {
    asm volatile("global_load_async_to_lds_b128 %0, %1, off"
                 :: "v"(lds_addr), "v"((unsigned long long)(uintptr_t)gaddr)
                 : "memory");
    asm volatile("global_load_async_to_lds_b128 %0, %1, off offset:16"
                 :: "v"(lds_addr), "v"((unsigned long long)(uintptr_t)gaddr)
                 : "memory");
}
static __device__ __forceinline__ void wait_async0() {
    asm volatile("s_wait_asynccnt 0x0" ::: "memory");
}

// ---------------------------------------------------------------------------
// GEMM: out[M,N] = act( (A[M,K] @ W[K,N] + bias) * scale ) (+ residual)
// A,W bf16 row-major; out f32 and/or bf16.  M%128==0, N%128==0, K%32==0.
// Block 256 thr (8 waves). Block tile 128x128, BK=32. Wave tile 64x32.
// ---------------------------------------------------------------------------
#define GBM 128
#define GBN 128
#define GBK 32
#define LDST 40   // padded LDS row stride (bf16 elems)

__global__ __launch_bounds__(256) void k_gemm(
    const bf16_t* __restrict__ A, const bf16_t* __restrict__ W,
    const float* __restrict__ bias, const float* __restrict__ resid,
    float* __restrict__ outf, bf16_t* __restrict__ outb,
    int M, int N, int K, float scale, int relu)
{
    __shared__ bf16_t At[GBM * LDST];
    __shared__ bf16_t Bt[GBN * LDST];

    const int tid    = threadIdx.x;
    const int lane   = tid & 31;
    const int wid    = tid >> 5;
    const int l16    = lane & 15;
    const int kbA    = (lane < 16) ? 0 : 16;   // A-frag K split (contiguous k = 16h + s, same as B)
    const int kbB    = (lane < 16) ? 0 : 16;   // B-frag K split
    const int rowoff = (lane < 16) ? 0 : 8;    // C/D-frag M split
    const int wm = wid >> 2, wn = wid & 3;
    const int m0 = blockIdx.y * GBM, n0 = blockIdx.x * GBN;

    v8f acc[4][2];
#pragma unroll
    for (int i = 0; i < 4; ++i)
#pragma unroll
        for (int j = 0; j < 2; ++j)
#pragma unroll
            for (int e = 0; e < 8; ++e) acc[i][j][e] = 0.f;

    const int arow = tid >> 1, aseg = tid & 1;   // A tile: 128 rows x 2 halves
    const int bkk  = tid >> 3, bng  = tid & 7;   // B tile: 32 k x 8 n-groups

    const unsigned at_lds = (unsigned)(uintptr_t)&At[arow * LDST + aseg * 16];

    for (int k0 = 0; k0 < K; k0 += GBK) {
        // A tile: async DMA global -> LDS (no VGPR round-trip)
        async_b128(at_lds, A + (size_t)(m0 + arow) * K + k0 + aseg * 16);

        // B tile -> LDS transposed [n][k] (VGPR-mediated reshape)
        {
            Frag t;
            const uint4* s = (const uint4*)(W + (size_t)(k0 + bkk) * N + n0 + bng * 16);
            t.u[0] = s[0]; t.u[1] = s[1];
#pragma unroll
            for (int j = 0; j < 16; ++j) Bt[(bng * 16 + j) * LDST + bkk] = t.h[j];
        }

        // Prefetch next K-tiles into cache while this tile computes
        if (k0 + GBK < K) {
            __builtin_prefetch(A + (size_t)(m0 + arow) * K + k0 + GBK + aseg * 16, 0, 0);
            __builtin_prefetch(W + (size_t)(k0 + GBK + bkk) * N + n0 + bng * 16, 0, 0);
        }

        wait_async0();          // our async LDS writes landed
        __syncthreads();        // + everyone's ds stores visible

        Frag af[4], bfm[2];
#pragma unroll
        for (int mi = 0; mi < 4; ++mi) {
            const bf16_t* p = &At[(wm * 64 + mi * 16 + l16) * LDST + kbA];
            af[mi].u[0] = *(const uint4*)p;
            af[mi].u[1] = *(const uint4*)(p + 8);
        }
#pragma unroll
        for (int ni = 0; ni < 2; ++ni) {
            const bf16_t* p = &Bt[(wn * 32 + ni * 16 + l16) * LDST + kbB];
            bfm[ni].u[0] = *(const uint4*)p;
            bfm[ni].u[1] = *(const uint4*)(p + 8);
        }
#pragma unroll
        for (int mi = 0; mi < 4; ++mi)
#pragma unroll
            for (int ni = 0; ni < 2; ++ni)
                acc[mi][ni] = wmma_bf16(af[mi].v, bfm[ni].v, acc[mi][ni]);
        __syncthreads();
    }

#pragma unroll
    for (int mi = 0; mi < 4; ++mi)
#pragma unroll
        for (int ni = 0; ni < 2; ++ni) {
            const int col = n0 + wn * 32 + ni * 16 + l16;
            const float bv = bias[col];
#pragma unroll
            for (int v = 0; v < 8; ++v) {
                const int row = m0 + wm * 64 + mi * 16 + rowoff + v;
                float val = (acc[mi][ni][v] + bv) * scale;
                if (relu) val = fmaxf(val, 0.f);
                const size_t idx = (size_t)row * N + col;
                if (resid) val += resid[idx];
                if (outf) outf[idx] = val;
                if (outb) outb[idx] = f2bf(val);
            }
        }
}

// ---------------------------------------------------------------------------
// Attention: one workgroup per (b, h, 64-query tile). Full 64 x Sk f32 score
// block in LDS (256 KB on CDNA5's 320 KB WGP LDS). Q pre-scaled by 1/sqrt(dk).
// Q,K: [B,S,D] bf16 (head h at column h*DK). Vt: [B,H,DK,S] bf16. O -> bf16.
// ---------------------------------------------------------------------------
__global__ __launch_bounds__(256) void k_attn(
    const bf16_t* __restrict__ Qb, const bf16_t* __restrict__ Kb,
    const bf16_t* __restrict__ Vt, const int* __restrict__ smask,
    bf16_t* __restrict__ Ob, int Sk, int causal)
{
    extern __shared__ float sc[];   // [64][Sk]

    const int tid    = threadIdx.x;
    const int lane   = tid & 31;
    const int wid    = tid >> 5;
    const int l16    = lane & 15;
    const int kbA    = (lane < 16) ? 0 : 16;
    const int kbB    = (lane < 16) ? 0 : 16;
    const int rowoff = (lane < 16) ? 0 : 8;
    const int q0 = blockIdx.x * 64;
    const int h  = blockIdx.y;
    const int b  = blockIdx.z;

    const bf16_t* Qh = Qb + (size_t)b * S_ * D_ + h * DK_;
    const bf16_t* Kh = Kb + (size_t)b * Sk * D_ + h * DK_;
    const bf16_t* Vh = Vt + ((size_t)(b * H_ + h)) * DK_ * Sk;

    // ---- Phase 1: scores = Q @ K^T (+mask) -> LDS -----------------------
    Frag qa[4][2];
#pragma unroll
    for (int mi = 0; mi < 4; ++mi)
#pragma unroll
        for (int ks = 0; ks < 2; ++ks) {
            const bf16_t* p = Qh + (size_t)(q0 + mi * 16 + l16) * D_ + ks * 32 + kbA;
            qa[mi][ks].u[0] = *(const uint4*)p;
            qa[mi][ks].u[1] = *(const uint4*)(p + 8);
        }
    const int nfmax = Sk >> 4;
    for (int nf = wid; nf < nfmax; nf += 8) {
        Frag bfr[2];
        const bf16_t* kp = Kh + (size_t)(nf * 16 + l16) * D_;   // key row = B column
#pragma unroll
        for (int ks = 0; ks < 2; ++ks) {
            bfr[ks].u[0] = *(const uint4*)(kp + ks * 32 + kbB);
            bfr[ks].u[1] = *(const uint4*)(kp + ks * 32 + kbB + 8);
        }
        const int key = nf * 16 + l16;
        const int mok = smask[b * Sk + key];
#pragma unroll
        for (int mi = 0; mi < 4; ++mi) {
            v8f a;
#pragma unroll
            for (int e = 0; e < 8; ++e) a[e] = 0.f;
            a = wmma_bf16(qa[mi][0].v, bfr[0].v, a);
            a = wmma_bf16(qa[mi][1].v, bfr[1].v, a);
#pragma unroll
            for (int v = 0; v < 8; ++v) {
                const int r = mi * 16 + rowoff + v;
                const bool ok = causal ? (key <= q0 + r) : (mok != 0);
                sc[r * Sk + key] = ok ? a[v] : -1e9f;
            }
        }
    }
    __syncthreads();

    // ---- Phase 2: row softmax (wave per row group) ----------------------
    for (int j = 0; j < 8; ++j) {
        const int r = wid * 8 + j;
        float* row = sc + (size_t)r * Sk;
        float m = -3.4e38f;
        for (int i = lane; i < Sk; i += 32) m = fmaxf(m, row[i]);
#pragma unroll
        for (int off = 16; off; off >>= 1) m = fmaxf(m, __shfl_xor(m, off, 32));
        float sum = 0.f;
        for (int i = lane; i < Sk; i += 32) { float e = __expf(row[i] - m); row[i] = e; sum += e; }
#pragma unroll
        for (int off = 16; off; off >>= 1) sum += __shfl_xor(sum, off, 32);
        const float inv = 1.f / sum;
        for (int i = lane; i < Sk; i += 32) row[i] *= inv;
    }
    __syncthreads();

    // ---- Phase 3: O = P @ V --------------------------------------------
    for (int p = 0; p < 2; ++p) {
        const int pair = wid * 2 + p;         // 16 (mi,ni) pairs over 8 waves
        const int mi = pair >> 2, ni = pair & 3;
        v8f acc;
#pragma unroll
        for (int e = 0; e < 8; ++e) acc[e] = 0.f;
        const float*  prow = sc + (size_t)(mi * 16 + l16) * Sk;
        const bf16_t* vrow = Vh + (size_t)(ni * 16 + l16) * Sk;   // dk row of V^T
        for (int ks = 0; ks < (Sk >> 5); ++ks) {
            Frag pa;
            const int ka = ks * 32 + kbA;
#pragma unroll
            for (int jj = 0; jj < 8; ++jj) {
                pa.h[jj]     = f2bf(prow[ka + jj]);
                pa.h[8 + jj] = f2bf(prow[ka + 8 + jj]);
            }
            Frag vb;
            const int kv = ks * 32 + kbB;
            vb.u[0] = *(const uint4*)(vrow + kv);
            vb.u[1] = *(const uint4*)(vrow + kv + 8);
            acc = wmma_bf16(pa.v, vb.v, acc);
        }
#pragma unroll
        for (int v = 0; v < 8; ++v) {
            const int qrow = q0 + mi * 16 + rowoff + v;
            const int dk   = ni * 16 + l16;
            Ob[((size_t)b * S_ + qrow) * D_ + h * DK_ + dk] = f2bf(acc[v]);
        }
    }
}

// ---------------------------------------------------------------------------
// Supporting kernels
// ---------------------------------------------------------------------------
__global__ void k_embed(const float* __restrict__ src,
                        float* __restrict__ xf, bf16_t* __restrict__ xb)
{
    const int idx = blockIdx.x * 256 + threadIdx.x;     // B*S*D exact
    const int d = idx % D_;
    const int s = (idx / D_) % S_;
    const float i2  = (float)(d & ~1);
    const float div = __expf(i2 * (-9.210340371976184f / (float)D_)); // ln(1e4)
    const float ang = (float)s * div;
    const float pe  = (d & 1) ? __cosf(ang) : __sinf(ang);
    const float v   = src[idx] + pe;
    xf[idx] = v;
    xb[idx] = f2bf(v);
}

__global__ void k_conv(const float* __restrict__ w, bf16_t* __restrict__ o, int n)
{
    const int i = blockIdx.x * 256 + threadIdx.x;
    if (i < n) o[i] = f2bf(w[i]);
}

// wq (H,D,DK) f32 -> W (D, H*DK) bf16 so Q/K/V become one DxD GEMM
__global__ void k_convh(const float* __restrict__ w, bf16_t* __restrict__ o)
{
    const int i = blockIdx.x * 256 + threadIdx.x;       // D*D exact
    const int d = i / D_, n = i % D_;
    const int h = n / DK_, k = n % DK_;
    o[(size_t)d * D_ + n] = f2bf(w[((size_t)h * D_ + d) * DK_ + k]);
}

// V [B,S,D] -> Vt [B,H,DK,S]
__global__ void k_vtrans(const bf16_t* __restrict__ v, bf16_t* __restrict__ vt)
{
    const int idx = blockIdx.x * 256 + threadIdx.x;     // B*S*D exact
    const int d = idx % D_;
    const int s = (idx / D_) % S_;
    const int b = idx / (D_ * S_);
    const int h = d / DK_, k = d % DK_;
    vt[(((size_t)b * H_ + h) * DK_ + k) * S_ + s] = v[idx];
}

__global__ __launch_bounds__(256) void k_ln(
    const float* __restrict__ x, const float* __restrict__ g,
    const float* __restrict__ bta, float* __restrict__ of, bf16_t* __restrict__ ob)
{
    __shared__ float red[256];
    const int row = blockIdx.x, tid = threadIdx.x;
    const float* xr = x + (size_t)row * D_;

    float s = 0.f;
    for (int i = tid; i < D_; i += 256) s += xr[i];
    red[tid] = s; __syncthreads();
    for (int o = 128; o; o >>= 1) { if (tid < o) red[tid] += red[tid + o]; __syncthreads(); }
    const float mu = red[0] / D_;
    __syncthreads();

    float vs = 0.f;
    for (int i = tid; i < D_; i += 256) { const float d = xr[i] - mu; vs += d * d; }
    red[tid] = vs; __syncthreads();
    for (int o = 128; o; o >>= 1) { if (tid < o) red[tid] += red[tid + o]; __syncthreads(); }
    const float inv = rsqrtf(red[0] / D_ + 1e-5f);

    for (int i = tid; i < D_; i += 256) {
        const float v = (xr[i] - mu) * inv * g[i] + bta[i];
        of[(size_t)row * D_ + i] = v;
        ob[(size_t)row * D_ + i] = f2bf(v);
    }
}

__global__ void k_copy(const float* __restrict__ in, float* __restrict__ out, int n)
{
    const int i = blockIdx.x * 256 + threadIdx.x;
    if (i < n) out[i] = in[i];
}

// ---------------------------------------------------------------------------
// Host orchestration
// ---------------------------------------------------------------------------
extern "C" void kernel_launch(void* const* d_in, const int* in_sizes, int n_in,
                              void* d_out, int out_size, void* d_ws, size_t ws_size,
                              hipStream_t stream)
{
    (void)in_sizes; (void)n_in; (void)out_size; (void)ws_size;

    // Workspace carve-out (~140 MB total)
    char* wp = (char*)d_ws;
    auto alloc = [&](size_t bytes) -> void* {
        void* p = wp; wp += (bytes + 255) & ~(size_t)255; return p;
    };
    float*  xe_f = (float*) alloc((size_t)M_ * D_ * 4);
    bf16_t* xe_b = (bf16_t*)alloc((size_t)M_ * D_ * 2);
    float*  yd_f = (float*) alloc((size_t)M_ * D_ * 4);
    bf16_t* yd_b = (bf16_t*)alloc((size_t)M_ * D_ * 2);
    float*  tmpf = (float*) alloc((size_t)M_ * D_ * 4);
    bf16_t* qb_  = (bf16_t*)alloc((size_t)M_ * D_ * 2);
    bf16_t* kb_  = (bf16_t*)alloc((size_t)M_ * D_ * 2);
    bf16_t* vb_  = (bf16_t*)alloc((size_t)M_ * D_ * 2);
    bf16_t* vt_  = (bf16_t*)alloc((size_t)M_ * D_ * 2);
    bf16_t* ao_  = (bf16_t*)alloc((size_t)M_ * D_ * 2);
    bf16_t* ffh_ = (bf16_t*)alloc((size_t)M_ * FF_ * 2);
    bf16_t* wbuf = (bf16_t*)alloc((size_t)FF_ * D_ * 2);

    auto F = [&](int i) { return (const float*)d_in[i]; };
    const int* smask = (const int*)d_in[2];

    // k_attn needs 256 KB dynamic LDS (CDNA5 WGP has 320 KB)
    const int attn_lds = 64 * S_ * (int)sizeof(float);
    (void)hipFuncSetAttribute((const void*)k_attn,
                              hipFuncAttributeMaxDynamicSharedMemorySize, attn_lds);

    auto gemm = [&](const bf16_t* A, const bf16_t* W, const float* bias,
                    const float* res, float* of, bf16_t* ob,
                    int M, int N, int K, float scale, int relu) {
        dim3 g(N / GBN, M / GBM);
        k_gemm<<<g, 256, 0, stream>>>(A, W, bias, res, of, ob, M, N, K, scale, relu);
    };
    auto convp = [&](const float* src, int n) {
        k_conv<<<(n + 255) / 256, 256, 0, stream>>>(src, wbuf, n);
    };
    auto convh = [&](const float* src) {
        k_convh<<<(D_ * D_) / 256, 256, 0, stream>>>(src, wbuf);
    };
    auto ln = [&](const float* in, const float* g, const float* b,
                  float* of, bf16_t* ob) {
        k_ln<<<M_, 256, 0, stream>>>(in, g, b, of, ob);
    };
    // Multi-head attention block: residual-summed f32 result -> tmpf
    auto mha = [&](const bf16_t* xq, const bf16_t* xkv, const float* res,
                   int pb, int causal) {
        convh(F(pb + 0)); gemm(xq,  wbuf, F(pb + 1), nullptr, nullptr, qb_, M_, D_, D_, 0.125f, 0);
        convh(F(pb + 2)); gemm(xkv, wbuf, F(pb + 3), nullptr, nullptr, kb_, M_, D_, D_, 1.f,    0);
        convh(F(pb + 4)); gemm(xkv, wbuf, F(pb + 5), nullptr, nullptr, vb_, M_, D_, D_, 1.f,    0);
        k_vtrans<<<(M_ * D_) / 256, 256, 0, stream>>>(vb_, vt_);
        dim3 ag(S_ / 64, H_, B_);
        k_attn<<<ag, 256, attn_lds, stream>>>(qb_, kb_, vt_, smask, ao_, S_, causal);
        convp(F(pb + 6), D_ * D_);
        gemm(ao_, wbuf, F(pb + 7), res, tmpf, nullptr, M_, D_, D_, 1.f, 0);
    };
    auto ffn = [&](const bf16_t* xb, const float* res,
                   const float* w1, const float* b1,
                   const float* w2, const float* b2) {
        convp(w1, D_ * FF_); gemm(xb,   wbuf, b1, nullptr, nullptr, ffh_, M_, FF_, D_, 1.f, 1);
        convp(w2, FF_ * D_); gemm(ffh_, wbuf, b2, res, tmpf, nullptr, M_, D_, FF_, 1.f, 0);
    };

    // Embedding + positional encoding
    k_embed<<<(M_ * D_) / 256, 256, 0, stream>>>(F(0), xe_f, xe_b);
    k_embed<<<(M_ * D_) / 256, 256, 0, stream>>>(F(1), yd_f, yd_b);

    // Encoder: 16 params/layer starting at index 4
    for (int l = 0; l < L_; ++l) {
        const int pb = 4 + l * 16;
        mha(xe_b, xe_b, xe_f, pb, /*causal=*/0);
        ln(tmpf, F(pb + 8), F(pb + 9), xe_f, xe_b);
        ffn(xe_b, xe_f, F(pb + 10), F(pb + 11), F(pb + 12), F(pb + 13));
        ln(tmpf, F(pb + 14), F(pb + 15), xe_f, xe_b);
    }

    // Decoder: 26 params/layer starting at index 4 + 16*L
    for (int l = 0; l < L_; ++l) {
        const int pb = 4 + 16 * L_ + l * 26;
        mha(yd_b, yd_b, yd_f, pb + 0, /*causal=*/1);          // masked self-attn
        ln(tmpf, F(pb + 16), F(pb + 17), yd_f, yd_b);
        mha(yd_b, xe_b, yd_f, pb + 8, /*causal=*/0);          // cross-attn on enc out
        ln(tmpf, F(pb + 18), F(pb + 19), yd_f, yd_b);
        ffn(yd_b, yd_f, F(pb + 22), F(pb + 23), F(pb + 24), F(pb + 25));
        ln(tmpf, F(pb + 20), F(pb + 21), yd_f, yd_b);
    }

    k_copy<<<(M_ * D_) / 256, 256, 0, stream>>>(yd_f, (float*)d_out, M_ * D_);
}
